// NAM_89739046683406
// MI455X (gfx1250) — compile-verified, matrix operand order checked by reading, not run on hardware
//
#include <hip/hip_runtime.h>
#include <hip/hip_bf16.h>
#include <math.h>

// ---------------------------------------------------------------------------
// NAM per-feature tiny MLP (B=32768, F=128, H=64), WMMA bf16 + TDM pipeline.
//   h1 = relu(x*w1+b1)        -> A fragment built directly in registers
//   h2 = relu(h1@w2 + b2)     -> v_wmma_f32_16x16x32_bf16, C seeded with b2
//   contrib = h2@w3           -> fused fma epilogue, butterfly N-reduction
//   out = [1-sigmoid, sigmoid](sum_f contrib + sum b3 + bias)
// w2 is pre-converted ONCE to bf16, transposed [g][h], padded to 144B rows in
// the workspace; the main kernel streams each 9216B feature tile into LDS with
// tensor_load_to_lds (TDM), double-buffered against compute (s_wait_tensorcnt).
// ---------------------------------------------------------------------------

typedef __bf16 bf16_t;
typedef __attribute__((ext_vector_type(16))) __bf16 v16bf;
typedef __attribute__((ext_vector_type(8)))  __bf16 v8bf;
typedef __attribute__((ext_vector_type(8)))  float  v8f;
typedef unsigned int u32;
typedef __attribute__((ext_vector_type(4))) u32 u32x4;
typedef __attribute__((ext_vector_type(8))) int  i32x8;
typedef __attribute__((ext_vector_type(4))) int  i32x4;

#define B_TOTAL      32768
#define F_DIM        128
#define H_DIM        64
#define WAVES_PER_WG 4
#define BLOCK        (WAVES_PER_WG * 32)
#define ROWS_PER_WG  (WAVES_PER_WG * 16)
#define WT_STRIDE    72                       // bf16 row stride: 144B, 16B-aligned
#define TILE_ELEMS   (H_DIM * WT_STRIDE)      // 4608 bf16 per feature tile
#define TILE_BYTES   (TILE_ELEMS * 2)         // 9216 B
#define WS_NEEDED    ((size_t)F_DIM * TILE_BYTES)

// ---- TDM: 1D DMA of one 9216B tile, global (bf16, pre-padded) -> LDS --------
__device__ __forceinline__ void tdm_load_tile(u32 lds_off, const bf16_t* gsrc) {
    unsigned long long ga = (unsigned long long)(uintptr_t)gsrc;
    u32x4 g0;
    g0[0] = 1u;                                            // count=1, user D#
    g0[1] = lds_off;                                       // lds_addr
    g0[2] = (u32)ga;                                       // global_addr[31:0]
    g0[3] = (u32)((ga >> 32) & 0x01FFFFFFu) | (2u << 30);  // [56:32] | type=2
    i32x8 g1;
    g1[0] = (2 << 16);                                     // data_size = 4B
    g1[1] = (TILE_BYTES / 4) << 16;                        // tensor_dim0 = 2304
    g1[2] = (1 << 16);                                     // tensor_dim1 = 1
    g1[3] = (TILE_BYTES / 4) << 16;                        // tile_dim0 = 2304
    g1[4] = 0;                                             // tile_dim1/2 unused
    g1[5] = (TILE_BYTES / 4);                              // tensor_dim0_stride
    g1[6] = 0;
    g1[7] = 0;
    i32x4 z4 = {0, 0, 0, 0};
    i32x8 z8 = {0, 0, 0, 0, 0, 0, 0, 0};
    __builtin_amdgcn_tensor_load_to_lds(g0, g1, z4, z4, z8, 0);
}

// ---- per-feature WMMA compute: A built in regs, B from LDS tile -------------
__device__ __forceinline__ void compute_feature(
    const bf16_t* __restrict__ tile, int f,
    int li, int koff, int myRow,
    const float* __restrict__ x,  const float* __restrict__ w1,
    const float* __restrict__ b1, const float* __restrict__ b2,
    const float* __restrict__ w3, float (&contrib)[8])
{
    const float xv = x[(size_t)myRow * F_DIM + f];
    const float* __restrict__ w1f = w1 + f * H_DIM;
    const float* __restrict__ b1f = b1 + f * H_DIM;
    v16bf a0, a1;
#pragma unroll
    for (int c = 0; c < 4; ++c) {       // c0->a0[0:8] c1->a0[8:16] c2->a1[0:8] c3->a1[8:16]
        const int base = koff + 16 * c;
        const float4 wa = *(const float4*)(w1f + base);
        const float4 wb = *(const float4*)(w1f + base + 4);
        const float4 ba = *(const float4*)(b1f + base);
        const float4 bb = *(const float4*)(b1f + base + 4);
        float hv[8];
        hv[0] = fmaf(xv, wa.x, ba.x); hv[1] = fmaf(xv, wa.y, ba.y);
        hv[2] = fmaf(xv, wa.z, ba.z); hv[3] = fmaf(xv, wa.w, ba.w);
        hv[4] = fmaf(xv, wb.x, bb.x); hv[5] = fmaf(xv, wb.y, bb.y);
        hv[6] = fmaf(xv, wb.z, bb.z); hv[7] = fmaf(xv, wb.w, bb.w);
#pragma unroll
        for (int e = 0; e < 8; ++e) {
            const bf16_t hb = (bf16_t)fmaxf(hv[e], 0.0f);
            if      (c == 0) a0[e]     = hb;
            else if (c == 1) a0[8 + e] = hb;
            else if (c == 2) a1[e]     = hb;
            else             a1[8 + e] = hb;
        }
    }

    const float* __restrict__ b2f = b2 + f * H_DIM;
    const float* __restrict__ w3f = w3 + f * H_DIM;
#pragma unroll
    for (int t = 0; t < 4; ++t) {
        const int g = t * 16 + li;                  // this lane's N column
        const bf16_t* rp = tile + g * WT_STRIDE;
        const v8bf B0lo = *(const v8bf*)(rp + koff);
        const v8bf B0hi = *(const v8bf*)(rp + koff + 16);
        const v8bf B1lo = *(const v8bf*)(rp + koff + 32);
        const v8bf B1hi = *(const v8bf*)(rp + koff + 48);
        v16bf Bf0, Bf1;
#pragma unroll
        for (int e = 0; e < 8; ++e) {
            Bf0[e] = B0lo[e]; Bf0[8 + e] = B0hi[e];
            Bf1[e] = B1lo[e]; Bf1[8 + e] = B1hi[e];
        }

        const float b2v = b2f[g];                   // C depends only on N -> splat
        v8f acc;
#pragma unroll
        for (int e = 0; e < 8; ++e) acc[e] = b2v;

        acc = __builtin_amdgcn_wmma_f32_16x16x32_bf16(
                  false, a0, false, Bf0, (short)0, acc, false, false);
        acc = __builtin_amdgcn_wmma_f32_16x16x32_bf16(
                  false, a1, false, Bf1, (short)0, acc, false, false);

        const float w3v = w3f[g];
#pragma unroll
        for (int e = 0; e < 8; ++e)
            contrib[e] = fmaf(fmaxf(acc[e], 0.0f), w3v, contrib[e]);
    }
}

// ---- final reduction + sigmoid output ---------------------------------------
__device__ __forceinline__ void finalize(
    float (&contrib)[8], int lane, int li, int hiHalf, int b0,
    const float* __restrict__ b3, const float* __restrict__ bias,
    float* __restrict__ out)
{
    float bsum = 0.0f;
    for (int f = lane; f < F_DIM; f += 32) bsum += b3[f];
#pragma unroll
    for (int m = 16; m >= 1; m >>= 1) bsum += __shfl_xor(bsum, m, 32);
    bsum += bias[0];

#pragma unroll
    for (int i = 0; i < 8; ++i) {
        float v = contrib[i];
#pragma unroll
        for (int m = 8; m >= 1; m >>= 1) v += __shfl_xor(v, m, 32);
        contrib[i] = v;
    }

    if (li == 0) {                 // lanes 0 / 16 hold M=0..7 / M=8..15 totals
#pragma unroll
        for (int i = 0; i < 8; ++i) {
            const int row   = b0 + hiHalf * 8 + i;
            const float lgt = contrib[i] + bsum;
            const float p   = 1.0f / (1.0f + __expf(-lgt));
            out[row * 2 + 0] = 1.0f - p;
            out[row * 2 + 1] = p;
        }
    }
}

// ---- one-time: w2 (F,H,H) f32 -> bf16, transposed [g][h], padded rows -------
__global__ __launch_bounds__(256) void preconvert_w2_kernel(
    const float* __restrict__ w2, bf16_t* __restrict__ wsT)
{
    const int f = blockIdx.x;
    const float* __restrict__ w2f = w2 + (size_t)f * (H_DIM * H_DIM);
    bf16_t* __restrict__ dst = wsT + (size_t)f * TILE_ELEMS;
    for (int i = threadIdx.x; i < TILE_ELEMS; i += 256) {
        const int g = i / WT_STRIDE;
        const int h = i - g * WT_STRIDE;
        dst[i] = (bf16_t)((h < H_DIM) ? w2f[h * H_DIM + g] : 0.0f);
    }
}

// ---- main kernel, TDM double-buffered ---------------------------------------
__global__ __launch_bounds__(BLOCK) void nam_mlp_tdm_kernel(
    const float* __restrict__ x,    const float* __restrict__ w1,
    const float* __restrict__ b1,   const float* __restrict__ b2,
    const float* __restrict__ w3,   const float* __restrict__ b3,
    const float* __restrict__ bias, const bf16_t* __restrict__ w2t,
    float* __restrict__ out)
{
    __shared__ __align__(16) bf16_t sWT[2][TILE_ELEMS];

    const int tid    = threadIdx.x;
    const int lane   = tid & 31;
    const int wave   = tid >> 5;
    const int li     = lane & 15;
    const int hiHalf = lane >> 4;
    const int koff   = hiHalf ? 8 : 0;
    const int b0     = blockIdx.x * ROWS_PER_WG + wave * 16;
    const int myRow  = b0 + li;

    float contrib[8];
#pragma unroll
    for (int i = 0; i < 8; ++i) contrib[i] = 0.0f;

    const u32 ldsOff[2] = { (u32)(uintptr_t)&sWT[0][0],
                            (u32)(uintptr_t)&sWT[1][0] };

    if (wave == 0) tdm_load_tile(ldsOff[0], w2t);      // prime buffer 0

    for (int f = 0; f < F_DIM; ++f) {
        if (wave == 0) __builtin_amdgcn_s_wait_tensorcnt(0);  // tile f landed
        __syncthreads();
        if (wave == 0 && f + 1 < F_DIM)                // overlap DMA(f+1)
            tdm_load_tile(ldsOff[(f + 1) & 1], w2t + (size_t)(f + 1) * TILE_ELEMS);

        compute_feature(sWT[f & 1], f, li, koff, myRow, x, w1, b1, b2, w3, contrib);
        __syncthreads();                                // readers done before reuse
    }

    finalize(contrib, lane, li, hiHalf, b0, b3, bias, out);
}

// ---- fallback (no workspace): self-contained LDS transpose ------------------
__global__ __launch_bounds__(BLOCK) void nam_mlp_fallback_kernel(
    const float* __restrict__ x,    const float* __restrict__ w1,
    const float* __restrict__ b1,   const float* __restrict__ w2,
    const float* __restrict__ b2,   const float* __restrict__ w3,
    const float* __restrict__ b3,   const float* __restrict__ bias,
    float* __restrict__ out)
{
    __shared__ __align__(16) bf16_t sWT[TILE_ELEMS];

    const int tid    = threadIdx.x;
    const int lane   = tid & 31;
    const int wave   = tid >> 5;
    const int li     = lane & 15;
    const int hiHalf = lane >> 4;
    const int koff   = hiHalf ? 8 : 0;
    const int b0     = blockIdx.x * ROWS_PER_WG + wave * 16;
    const int myRow  = b0 + li;

    float contrib[8];
#pragma unroll
    for (int i = 0; i < 8; ++i) contrib[i] = 0.0f;

    for (int f = 0; f < F_DIM; ++f) {
        __syncthreads();
        const float* __restrict__ w2f = w2 + (size_t)f * (H_DIM * H_DIM);
        for (int i4 = tid * 4; i4 < H_DIM * H_DIM; i4 += BLOCK * 4) {
            const float4 v = *(const float4*)(w2f + i4);
            const int h = i4 >> 6, g = i4 & 63;
            sWT[(g + 0) * WT_STRIDE + h] = (bf16_t)v.x;
            sWT[(g + 1) * WT_STRIDE + h] = (bf16_t)v.y;
            sWT[(g + 2) * WT_STRIDE + h] = (bf16_t)v.z;
            sWT[(g + 3) * WT_STRIDE + h] = (bf16_t)v.w;
        }
        if (f + 1 < F_DIM)
            __builtin_prefetch(w2 + (size_t)(f + 1) * (H_DIM * H_DIM) + tid * 32, 0, 1);
        __syncthreads();

        compute_feature(sWT, f, li, koff, myRow, x, w1, b1, b2, w3, contrib);
    }

    finalize(contrib, lane, li, hiHalf, b0, b3, bias, out);
}

extern "C" void kernel_launch(void* const* d_in, const int* in_sizes, int n_in,
                              void* d_out, int out_size, void* d_ws, size_t ws_size,
                              hipStream_t stream) {
    (void)in_sizes; (void)n_in; (void)out_size;
    const float* x    = (const float*)d_in[0];
    const float* w1   = (const float*)d_in[1];
    const float* b1   = (const float*)d_in[2];
    const float* w2   = (const float*)d_in[3];
    const float* b2   = (const float*)d_in[4];
    const float* w3   = (const float*)d_in[5];
    const float* b3   = (const float*)d_in[6];
    const float* bias = (const float*)d_in[7];
    float* out = (float*)d_out;

    dim3 grid(B_TOTAL / ROWS_PER_WG);       // 512 workgroups x 4 waves

    if (ws_size >= WS_NEEDED) {
        bf16_t* w2t = (bf16_t*)d_ws;
        preconvert_w2_kernel<<<dim3(F_DIM), 256, 0, stream>>>(w2, w2t);
        nam_mlp_tdm_kernel<<<grid, BLOCK, 0, stream>>>(
            x, w1, b1, b2, w3, b3, bias, w2t, out);
    } else {
        nam_mlp_fallback_kernel<<<grid, BLOCK, 0, stream>>>(
            x, w1, b1, w2, b2, w3, b3, bias, out);
    }
}
